// AdvancedGNNModel_16776142259102
// MI455X (gfx1250) — matrix operand order repaired and checked by reading, not compiled, not run on hardware
//
#include <hip/hip_runtime.h>
#include <hip/hip_bf16.h>
#include <stdint.h>

typedef __attribute__((ext_vector_type(16))) _Float16     v16h;
typedef __attribute__((ext_vector_type(8)))  float        v8f;
typedef __attribute__((ext_vector_type(4)))  unsigned int v4u;
typedef __attribute__((ext_vector_type(8)))  int          v8i;
typedef __attribute__((ext_vector_type(4)))  int          v4i;

#define F_IN 128
#define HC   128
#define NH   4
#define CH   32

// ---------- float atomic max via sign-aware integer atomics ----------
__device__ __forceinline__ void atomicMaxF(float* addr, float val) {
    if (!(val < 0.0f)) {
        atomicMax((int*)addr, __float_as_int(val));
    } else {
        atomicMin((unsigned int*)addr, __float_as_uint(val));
    }
}

// ---------- f32 -> f16 conversion ----------
__global__ void cvt_f16_kernel(const float* __restrict__ w,
                               _Float16* __restrict__ wh, int n) {
    int i = blockIdx.x * blockDim.x + threadIdx.x;
    if (i < n) wh[i] = (_Float16)w[i];
}

#if __has_builtin(__builtin_amdgcn_tensor_load_to_lds)
#define HAVE_TDM 1
// Issue a TDM load of a 2-D f16 tile [tile1 x 128] from a row-major tensor
// (row stride 128 halves) into LDS with 16B padding every 256B (row stride
// 272B in LDS == [][136] _Float16 layout). Rows beyond dim1 read as zero.
__device__ __forceinline__ void tdm_load_tile_f16(uint32_t lds_addr, uint64_t gaddr,
                                                  int dim1, int tile1) {
    v4u g0;
    g0[0] = 1u;                                   // count=1 valid descriptor
    g0[1] = lds_addr;                             // LDS byte address
    g0[2] = (uint32_t)gaddr;                      // global_addr[31:0]
    g0[3] = (uint32_t)((gaddr >> 32) & 0x01ffffffu) | 0x80000000u; // addr[56:32] | type=2
    v8i g1;
    // data_size=1 (2B), pad_enable, pad_interval=5 (64 DW), pad_amount=3 (4 DW)
    g1[0] = (1 << 16) | (1 << 20) | (5 << 22) | (3 << 25);
    g1[1] = (F_IN << 16);                         // tensor_dim0 = 128
    g1[2] = (dim1 & 0xffff) << 16;                // tensor_dim1[15:0]
    g1[3] = ((dim1 >> 16) & 0xffff) | (F_IN << 16); // tensor_dim1[31:16] | tile_dim0=128
    g1[4] = (tile1 & 0xffff);                     // tile_dim1 (tile_dim2=0)
    g1[5] = F_IN;                                 // tensor_dim0_stride = 128
    g1[6] = 0;
    g1[7] = 0;
    v4i gz = {0, 0, 0, 0};
#if __clang_major__ >= 23
    v8i gz8 = {0, 0, 0, 0, 0, 0, 0, 0};
    __builtin_amdgcn_tensor_load_to_lds(g0, g1, gz, gz, gz8, 0);
#else
    __builtin_amdgcn_tensor_load_to_lds(g0, g1, gz, gz, 0);
#endif
}
#else
#define HAVE_TDM 0
#endif

// ---------- WMMA GEMM: C[N,OC] = Ah[N,128] * Wh[OC,128]^T (+bias) ----------
// 256 threads = 8 waves; block computes a 128-row x OC tile. Tiles arrive in
// LDS via the Tensor Data Mover (wave 0 issues 2 descriptors, waits on
// TENSORcnt, then workgroup barrier releases all waves).
template <int OC, bool ADD_BIAS>
__global__ __launch_bounds__(256) void gemm_wmma_kernel(
    const _Float16* __restrict__ Ah,    // [N,128] f16
    const _Float16* __restrict__ Wh,    // [OC,128] f16
    const float* __restrict__ bias,     // [OC] or null
    float* __restrict__ Cout,           // [N,OC] f32
    int N)
{
    __shared__ _Float16 As[128][F_IN + 8];
    __shared__ _Float16 Bs[OC][F_IN + 8];

    const int tid = threadIdx.x;
    const int rowBase = blockIdx.x * 128;

#if HAVE_TDM
    if (tid < 32) {   // wave 0 issues both tile DMAs (TDM ignores EXEC)
        tdm_load_tile_f16((uint32_t)(uintptr_t)&As[0][0],
                          (uint64_t)(uintptr_t)(Ah + (size_t)rowBase * F_IN),
                          N - rowBase, 128);
        tdm_load_tile_f16((uint32_t)(uintptr_t)&Bs[0][0],
                          (uint64_t)(uintptr_t)Wh, OC, OC);
        __builtin_amdgcn_s_wait_tensorcnt(0);
    }
#else
    // Fallback: cooperative 16B copies
    for (int i = tid; i < 128 * (F_IN / 8); i += 256) {
        int r = i >> 4, kc = (i & 15) << 3;
        uint4 v = make_uint4(0u, 0u, 0u, 0u);
        if (rowBase + r < N)
            v = *(const uint4*)(Ah + (size_t)(rowBase + r) * F_IN + kc);
        *(uint4*)&As[r][kc] = v;
    }
    for (int i = tid; i < OC * (F_IN / 8); i += 256) {
        int r = i >> 4, kc = (i & 15) << 3;
        *(uint4*)&Bs[r][kc] = *(const uint4*)(Wh + (size_t)r * F_IN + kc);
    }
#endif
    __syncthreads();

    const int wave = tid >> 5;
    const int lane = tid & 31;
    const int m15  = lane & 15;
    const int hi   = lane >> 4;
    const int arow = (wave << 4) + m15;
    const int khA  = hi << 3;           // A frag: K interleave of 8
    const int khB  = hi << 4;           // unused; B frag uses khA (interleave of 8)

    // Hoist all A fragments (reused across every column tile)
    v16h afr[F_IN / 32];
#pragma unroll
    for (int kk = 0; kk < F_IN / 32; ++kk) {
        const int kb = kk << 5;
#pragma unroll
        for (int i = 0; i < 8; ++i) {
            int ka = kb + ((i & 4) << 2) + khA + ((i & 3) << 1);
            afr[kk][2 * i]     = As[arow][ka];
            afr[kk][2 * i + 1] = As[arow][ka + 1];
        }
    }

    const bool fullTile = (rowBase + 128 <= N);
    const int gr0 = rowBase + (wave << 4) + (hi << 3);

#pragma unroll
    for (int jt = 0; jt < OC / 16; ++jt) {
        v8f acc = {};
        const int bcol = (jt << 4) + m15;
#pragma unroll
        for (int kk = 0; kk < F_IN / 32; ++kk) {
            v16h b;
            const int kb = kk << 5;
#pragma unroll
            for (int i = 0; i < 8; ++i) {
                int kbb = kb + ((i & 4) << 2) + khA + ((i & 3) << 1);
                b[2 * i]     = Bs[bcol][kbb];
                b[2 * i + 1] = Bs[bcol][kbb + 1];
            }
            acc = __builtin_amdgcn_wmma_f32_16x16x32_f16(
                false, afr[kk], false, b, (short)0, acc, false, false);
        }
        float bv = 0.f;
        if (ADD_BIAS) bv = bias[bcol];
        if (fullTile) {
#pragma unroll
            for (int r = 0; r < 8; ++r) {
                float v = acc[r];
                if (ADD_BIAS) v += bv;
                Cout[(size_t)(gr0 + r) * OC + bcol] = v;
            }
        } else {
#pragma unroll
            for (int r = 0; r < 8; ++r) {
                if (gr0 + r < N) {
                    float v = acc[r];
                    if (ADD_BIAS) v += bv;
                    Cout[(size_t)(gr0 + r) * OC + bcol] = v;
                }
            }
        }
    }
}

// ---------- per-(node,head) attention logits ----------
__global__ void alpha_kernel(const float* __restrict__ h,
                             const float* __restrict__ a_src,
                             const float* __restrict__ a_dst,
                             float* __restrict__ as_out,
                             float* __restrict__ ad_out, int N)
{
    int i = blockIdx.x * blockDim.x + threadIdx.x;
    if (i >= N * NH) return;
    int n = i >> 2, hh = i & 3;
    const float* hp = h + (size_t)n * HC + hh * CH;
    const float* ps = a_src + hh * CH;
    const float* pd = a_dst + hh * CH;
    float s = 0.f, d = 0.f;
#pragma unroll
    for (int c = 0; c < CH; ++c) {
        float v = hp[c];
        s += v * ps[c];
        d += v * pd[c];
    }
    as_out[i] = s;
    ad_out[i] = d;
}

__global__ void init_kernel(float* __restrict__ agg, float* __restrict__ m,
                            float* __restrict__ s, int N) {
    int i = blockIdx.x * blockDim.x + threadIdx.x;
    if (i < N * HC) agg[i] = 0.f;
    if (i < N * NH) { m[i] = -3.0e38f; s[i] = 0.f; }
}

__device__ __forceinline__ void edge_ids(int e, int E, const int* __restrict__ src,
                                         const int* __restrict__ dst, int& s_, int& d_) {
    if (e < E) { s_ = src[e]; d_ = dst[e]; }
    else       { s_ = e - E;  d_ = e - E; }   // appended self loops
}

__global__ void edge_max_kernel(const int* __restrict__ src, const int* __restrict__ dst,
                                const float* __restrict__ as, const float* __restrict__ ad,
                                float* __restrict__ m, int E, int EP)
{
    int i = blockIdx.x * blockDim.x + threadIdx.x;
    if (i >= EP * NH) return;
    int e = i >> 2, hh = i & 3, s_, d_;
    edge_ids(e, E, src, dst, s_, d_);
    float v = as[s_ * NH + hh] + ad[d_ * NH + hh];
    v = (v > 0.f) ? v : 0.2f * v;            // leaky_relu
    atomicMaxF(&m[d_ * NH + hh], v);
}

__global__ void edge_sum_kernel(const int* __restrict__ src, const int* __restrict__ dst,
                                const float* __restrict__ as, const float* __restrict__ ad,
                                const float* __restrict__ m, float* __restrict__ ssum,
                                int E, int EP)
{
    int i = blockIdx.x * blockDim.x + threadIdx.x;
    if (i >= EP * NH) return;
    int e = i >> 2, hh = i & 3, s_, d_;
    edge_ids(e, E, src, dst, s_, d_);
    float v = as[s_ * NH + hh] + ad[d_ * NH + hh];
    v = (v > 0.f) ? v : 0.2f * v;
    float w = __expf(v - m[d_ * NH + hh]);
    atomicAdd(&ssum[d_ * NH + hh], w);
}

// one wave32 per edge; each lane carries 4 channels (b128 load, 4 f32 L2 atomics)
__global__ __launch_bounds__(256) void edge_agg_kernel(
    const int* __restrict__ src, const int* __restrict__ dst,
    const float* __restrict__ as, const float* __restrict__ ad,
    const float* __restrict__ m, const float* __restrict__ ssum,
    const float* __restrict__ h, float* __restrict__ agg,
    int E, int EP)
{
    int gid = blockIdx.x * blockDim.x + threadIdx.x;
    int e = gid >> 5;
    if (e >= EP) return;
    int lane = gid & 31;
    int s_, d_;
    edge_ids(e, E, src, dst, s_, d_);
    int hh = lane >> 3;
    float v = as[s_ * NH + hh] + ad[d_ * NH + hh];
    v = (v > 0.f) ? v : 0.2f * v;
    float w = __expf(v - m[d_ * NH + hh]);
    float alpha = w / (ssum[d_ * NH + hh] + 1e-16f);
    int c0 = lane << 2;
    float4 hv = *(const float4*)(h + (size_t)s_ * HC + c0);
    float* op = agg + (size_t)d_ * HC + c0;
    atomicAdd(op + 0, alpha * hv.x);
    atomicAdd(op + 1, alpha * hv.y);
    atomicAdd(op + 2, alpha * hv.z);
    atomicAdd(op + 3, alpha * hv.w);
}

// bias + relu, emit next-layer activations directly in f16
__global__ void bias_relu_f16_kernel(const float* __restrict__ agg,
                                     const float* __restrict__ bias,
                                     _Float16* __restrict__ xout, int N) {
    int i = blockIdx.x * blockDim.x + threadIdx.x;
    if (i >= N * HC) return;
    float v = agg[i] + bias[i & (HC - 1)];
    xout[i] = (_Float16)(v > 0.f ? v : 0.f);
}

extern "C" void kernel_launch(void* const* d_in, const int* in_sizes, int n_in,
                              void* d_out, int out_size, void* d_ws, size_t ws_size,
                              hipStream_t stream) {
    const float* x   = (const float*)d_in[0];
    const float* Ws  = (const float*)d_in[1];
    const float* asr = (const float*)d_in[2];
    const float* ads = (const float*)d_in[3];
    const float* cb  = (const float*)d_in[4];
    const float* Wf  = (const float*)d_in[5];
    const float* bf  = (const float*)d_in[6];
    const int*   ei  = (const int*)d_in[7];

    const int N  = in_sizes[0] / F_IN;
    const int L  = in_sizes[1] / (HC * F_IN);
    const int E  = in_sizes[7] / 2;
    const int EP = E + N;
    const int* src = ei;
    const int* dst = ei + E;

    char* base = (char*)d_ws;
    size_t off = 0;
    auto carve = [&](size_t bytes) -> void* {
        void* p = base + off;
        off = (off + bytes + 255) & ~(size_t)255;
        return p;
    };
    float*    h    = (float*)carve((size_t)N * HC * 4);
    float*    agg  = (float*)carve((size_t)N * HC * 4);
    _Float16* xh   = (_Float16*)carve((size_t)N * HC * 2);
    float*    al_s = (float*)carve((size_t)N * NH * 4);
    float*    al_d = (float*)carve((size_t)N * NH * 4);
    float*    mmax = (float*)carve((size_t)N * NH * 4);
    float*    ssum = (float*)carve((size_t)N * NH * 4);
    _Float16* Wh   = (_Float16*)carve((size_t)HC * F_IN * 2);

    const dim3 b256(256);
    const int gGemm = (N + 127) / 128;

    // layer-0 activations -> f16 staging
    cvt_f16_kernel<<<(N * HC + 255) / 256, b256, 0, stream>>>(x, xh, N * HC);

    for (int l = 0; l < L; ++l) {
        cvt_f16_kernel<<<(HC * F_IN + 255) / 256, b256, 0, stream>>>(
            Ws + (size_t)l * HC * F_IN, Wh, HC * F_IN);
        gemm_wmma_kernel<HC, false><<<gGemm, b256, 0, stream>>>(xh, Wh, nullptr, h, N);
        alpha_kernel<<<(N * NH + 255) / 256, b256, 0, stream>>>(
            h, asr + l * NH * CH, ads + l * NH * CH, al_s, al_d, N);
        init_kernel<<<(N * HC + 255) / 256, b256, 0, stream>>>(agg, mmax, ssum, N);
        edge_max_kernel<<<(EP * NH + 255) / 256, b256, 0, stream>>>(
            src, dst, al_s, al_d, mmax, E, EP);
        edge_sum_kernel<<<(EP * NH + 255) / 256, b256, 0, stream>>>(
            src, dst, al_s, al_d, mmax, ssum, E, EP);
        edge_agg_kernel<<<(int)(((size_t)EP * 32 + 255) / 256), b256, 0, stream>>>(
            src, dst, al_s, al_d, mmax, ssum, h, agg, E, EP);
        bias_relu_f16_kernel<<<(N * HC + 255) / 256, b256, 0, stream>>>(
            agg, cb + l * HC, xh, N);
    }
    // final linear head: [N,128] @ [32,128]^T + bf
    cvt_f16_kernel<<<(32 * F_IN + 255) / 256, b256, 0, stream>>>(Wf, Wh, 32 * F_IN);
    gemm_wmma_kernel<32, true><<<gGemm, b256, 0, stream>>>(xh, Wh, bf, (float*)d_out, N);
}
